// EntityComposer_7859790151956
// MI455X (gfx1250) — matrix operand order repaired and checked by reading, not compiled, not run on hardware
//
#include <hip/hip_runtime.h>
#include <hip/hip_bf16.h>
#include <math.h>

// Problem dims
#define BSZ   128
#define NE    32
#define NATTR 8
#define TE    8
#define TA    4
#define WD    300
#define HD    1024
#define G     (BSZ*NE)      // 4096 entity seqs / graphs
#define GA    (G*NATTR)     // 32768 attribute seqs
#define KP    320           // WD padded to mult of 32
#define KC_W  (KP/32)       // 10 K-chunks for D=300 GEMMs
#define NG    900           // 3*WD gate width
#define NGT   57            // ceil(900/16) N-tiles
#define HDT   64            // 1024/16 N-tiles
#define KC_H  (HD/32)       // 32 K-chunks for D=1024 GEMM
#define NNODE 9             // 1 entity + 8 attribute nodes

typedef __attribute__((ext_vector_type(16))) __bf16 v16bf;
typedef __attribute__((ext_vector_type(8)))  float  v8f;

__device__ __forceinline__ unsigned short f2bf(float f) {
  union { float f; unsigned u; } v; v.f = f;
  unsigned u = v.u + 0x7fffu + ((v.u >> 16) & 1u);   // round-to-nearest-even
  return (unsigned short)(u >> 16);
}

// A-fragment (16x32 bf16) from LDS, per ISA 7.12.2: lane&15 = row M,
// lane>=16 picks K octets {8..15,24..31}, else {0..7,16..23}.
__device__ __forceinline__ v16bf load_a_frag(const unsigned short* sA, int ldk,
                                             int lane, int kc) {
  const int row = lane & 15;
  const int hi  = (lane >> 4) & 1;
  const __bf16* p = (const __bf16*)(sA + row * ldk + kc * 32 + hi * 8);
  v16bf a;
#pragma unroll
  for (int i = 0; i < 8; ++i) { a[i] = p[i]; a[i + 8] = p[i + 16]; }
  return a;
}

// B-fragment: pre-swizzled in workspace; each lane reads 16 contiguous bf16.
__device__ __forceinline__ v16bf load_b_frag(const unsigned short* pk, int frag,
                                             int lane) {
  return *(const v16bf*)(pk + ((size_t)frag << 9) + (lane << 4));
}

// ---------------------------------------------------------------------------
// Pack fp32 weight matrix W[rows][cols] into bf16 WMMA-B fragments.
// B = W^T (we compute X @ W^T): frag(nt,kc), lane l, elem i ->
//   N = l&15, K = (l>=16?8:0)+i+(i>=8?8:0) ; weight element W[nt*16+N][kc*32+K].
// ---------------------------------------------------------------------------
__global__ __launch_bounds__(256)
void pack_b_kernel(const float* __restrict__ W, unsigned short* __restrict__ out,
                   int rows, int cols, int ntiles, int kchunks) {
  int idx = blockIdx.x * 256 + threadIdx.x;
  int total = ntiles * kchunks * 512;
  if (idx >= total) return;
  int i    = idx & 15;
  int lane = (idx >> 4) & 31;
  int frag = idx >> 9;
  int kc = frag % kchunks, nt = frag / kchunks;
  int n = lane & 15;
  int k = ((lane >> 4) ? 8 : 0) + i + (i >= 8 ? 8 : 0);
  int wr = nt * 16 + n;
  int wc = kc * 32 + k;
  float f = (wr < rows && wc < cols) ? W[(size_t)wr * cols + wc] : 0.f;
  out[idx] = f2bf(f);
}

// ---------------------------------------------------------------------------
// Fused embedding-gather + GRU. 16 sequences per workgroup, 4 waves.
// Recurrence lives entirely in LDS; dual WMMA GEMM per timestep.
// ---------------------------------------------------------------------------
__global__ __launch_bounds__(128)
void gru_wmma_kernel(const int* __restrict__ tokens, const int* __restrict__ lengths,
                     const float* __restrict__ emb,
                     const unsigned short* __restrict__ wih_pk,
                     const unsigned short* __restrict__ whh_pk,
                     const float* __restrict__ bih, const float* __restrict__ bhh,
                     unsigned short* __restrict__ h_out, int T) {
  __shared__ unsigned short sX[16 * KP];   // x_t   (bf16, K-padded)
  __shared__ unsigned short sH[16 * KP];   // h     (bf16, K-padded)
  __shared__ float sXg[16 * 912];          // r,z: xg+hg summed; n: xg only
  __shared__ float sHn[16 * 304];          // n-gate hidden preact (hn)
  __shared__ float sHf[16 * 304];          // fp32 master h
  __shared__ int   sLen[16];

  const int tid  = threadIdx.x;
  const int lane = tid & 31;
  const int wv   = tid >> 5;
  const int seq0 = blockIdx.x * 16;

  for (int i = tid; i < 16 * KP; i += 128) { sX[i] = 0; sH[i] = 0; }
  for (int i = tid; i < 16 * 304; i += 128) sHf[i] = 0.f;
  if (tid < 16) { int L = lengths[seq0 + tid]; sLen[tid] = L < 1 ? 1 : L; }
  __syncthreads();

  for (int t = 0; t < T; ++t) {
    // gather token embedding rows -> bf16 LDS
    for (int i = tid; i < 16 * WD; i += 128) {
      int m = i / WD, c = i - m * WD;
      int tok = tokens[(seq0 + m) * T + t];
      sX[m * KP + c] = f2bf(emb[(size_t)tok * WD + c]);
    }
    __syncthreads();

    // dual GEMM: xg = x_t @ Wih^T ; hg = h @ Whh^T   (M=16, N=912, K=320)
    for (int nt = wv; nt < NGT; nt += 4) {
      v8f ax, ah;
#pragma unroll
      for (int j = 0; j < 8; ++j) { ax[j] = 0.f; ah[j] = 0.f; }
#pragma unroll
      for (int kc = 0; kc < KC_W; ++kc) {
        v16bf a0 = load_a_frag(sX, KP, lane, kc);
        v16bf b0 = load_b_frag(wih_pk, nt * KC_W + kc, lane);
        ax = __builtin_amdgcn_wmma_f32_16x16x32_bf16(false, a0, false, b0,
                                                     (short)0, ax, false, false);
        v16bf a1 = load_a_frag(sH, KP, lane, kc);
        v16bf b1 = load_b_frag(whh_pk, nt * KC_W + kc, lane);
        ah = __builtin_amdgcn_wmma_f32_16x16x32_bf16(false, a1, false, b1,
                                                     (short)0, ah, false, false);
      }
      const int col = nt * 16 + (lane & 15);
      if (col < NG) {
        const int rbase = (lane >> 4) * 8;
#pragma unroll
        for (int j = 0; j < 8; ++j) {
          int row = rbase + j;
          float xg = ax[j] + bih[col];
          float hg = ah[j] + bhh[col];
          if (col < 2 * WD) sXg[row * 912 + col] = xg + hg;        // r,z sums
          else { sXg[row * 912 + col] = xg;                        // xn
                 sHn[row * 304 + (col - 2 * WD)] = hg; }           // hn
        }
      }
    }
    __syncthreads();

    // gate math (fp32 VALU), freeze when t >= length
    for (int i = tid; i < 16 * WD; i += 128) {
      int m = i / WD, c = i - m * WD;
      if (t < sLen[m]) {
        float r = 1.f / (1.f + __expf(-sXg[m * 912 + c]));
        float z = 1.f / (1.f + __expf(-sXg[m * 912 + WD + c]));
        float n = tanhf(sXg[m * 912 + 2 * WD + c] + r * sHn[m * 304 + c]);
        float h = sHf[m * 304 + c];
        float hn = (1.f - z) * n + z * h;
        sHf[m * 304 + c] = hn;
        sH[m * KP + c] = f2bf(hn);
      }
    }
    __syncthreads();
  }

  for (int i = tid; i < 16 * KP; i += 128) {
    int m = i / KP, c = i - m * KP;
    h_out[(size_t)(seq0 + m) * KP + c] = sH[m * KP + c];
  }
}

// ---------------------------------------------------------------------------
// FC + ReLU: [16 rows x 320] @ Wfc^T -> bf16 rows scattered into node tensor.
// mode 0: entity rows (node 0); mode 1: attribute rows (nodes 1..8).
// ---------------------------------------------------------------------------
__global__ __launch_bounds__(256)
void fc_relu_wmma_kernel(const unsigned short* __restrict__ h_bf,
                         const unsigned short* __restrict__ wfc_pk,
                         const float* __restrict__ bfc,
                         unsigned short* __restrict__ nodes_bf, int mode) {
  __shared__ unsigned short sA[16 * KP];
  const int tid = threadIdx.x, lane = tid & 31, wv = tid >> 5;  // 8 waves
  const int r0 = blockIdx.x * 16;
  for (int i = tid; i < 16 * KP; i += 256) sA[i] = h_bf[(size_t)r0 * KP + i];
  __syncthreads();

  v8f acc[8];
#pragma unroll
  for (int q = 0; q < 8; ++q)
#pragma unroll
    for (int j = 0; j < 8; ++j) acc[q][j] = 0.f;

  for (int kc = 0; kc < KC_W; ++kc) {
    v16bf a = load_a_frag(sA, KP, lane, kc);     // shared across 8 N-tiles
#pragma unroll
    for (int q = 0; q < 8; ++q) {
      int nt = wv * 8 + q;
      v16bf b = load_b_frag(wfc_pk, nt * KC_W + kc, lane);
      acc[q] = __builtin_amdgcn_wmma_f32_16x16x32_bf16(false, a, false, b,
                                                       (short)0, acc[q], false, false);
    }
  }
  const int rbase = (lane >> 4) * 8;
#pragma unroll
  for (int q = 0; q < 8; ++q) {
    int nt = wv * 8 + q;
    int col = nt * 16 + (lane & 15);
    float bb = bfc[col];
#pragma unroll
    for (int j = 0; j < 8; ++j) {
      int arow = r0 + rbase + j;
      float v = acc[q][j] + bb;
      v = v > 0.f ? v : 0.f;
      size_t nrow = (mode == 0) ? (size_t)arow * NNODE
                                : (size_t)(arow >> 3) * NNODE + 1 + (arow & 7);
      nodes_bf[nrow * HD + col] = f2bf(v);
    }
  }
}

// ---------------------------------------------------------------------------
// GAT projection: hg = nodes @ Wg^T   (M=36864, N=1024, K=1024), fp32 out.
// ---------------------------------------------------------------------------
__global__ __launch_bounds__(256)
void gat_gemm_wmma_kernel(const unsigned short* __restrict__ nodes_bf,
                          const unsigned short* __restrict__ wg_pk,
                          float* __restrict__ hg) {
  __shared__ unsigned short sA[16 * HD];
  const int tid = threadIdx.x, lane = tid & 31, wv = tid >> 5;
  const int r0 = blockIdx.x * 16;
  for (int i = tid; i < 16 * HD; i += 256) sA[i] = nodes_bf[(size_t)r0 * HD + i];
  __syncthreads();

  v8f acc[8];
#pragma unroll
  for (int q = 0; q < 8; ++q)
#pragma unroll
    for (int j = 0; j < 8; ++j) acc[q][j] = 0.f;

  for (int kc = 0; kc < KC_H; ++kc) {
    v16bf a = load_a_frag(sA, HD, lane, kc);
#pragma unroll
    for (int q = 0; q < 8; ++q) {
      int nt = wv * 8 + q;
      __builtin_prefetch(wg_pk + ((size_t)(nt * KC_H + kc + 1) << 9), 0, 1);
      v16bf b = load_b_frag(wg_pk, nt * KC_H + kc, lane);
      acc[q] = __builtin_amdgcn_wmma_f32_16x16x32_bf16(false, a, false, b,
                                                       (short)0, acc[q], false, false);
    }
  }
  const int rbase = (lane >> 4) * 8;
#pragma unroll
  for (int q = 0; q < 8; ++q) {
    int nt = wv * 8 + q, col = nt * 16 + (lane & 15);
#pragma unroll
    for (int j = 0; j < 8; ++j)
      hg[(size_t)(r0 + rbase + j) * HD + col] = acc[q][j];
  }
}

// ---------------------------------------------------------------------------
// Per-graph masked attention readout (only node 0's output is needed).
// ---------------------------------------------------------------------------
__global__ __launch_bounds__(256)
void gat_attn_kernel(const float* __restrict__ hg, const float* __restrict__ a_src,
                     const float* __restrict__ a_dst,
                     const unsigned char* __restrict__ attr_mask,
                     float* __restrict__ out) {
  __shared__ float sDot[10];     // 0..8 = ed[j]; 9 = es[0]
  __shared__ float sAlpha[NNODE];
  const int g = blockIdx.x;
  const int tid = threadIdx.x, lane = tid & 31, wv = tid >> 5;
  const float* base = hg + (size_t)g * NNODE * HD;

  for (int dp = wv; dp < 10; dp += 8) {
    const float* vec  = (dp < 9) ? (base + (size_t)dp * HD) : base;
    const float* coef = (dp < 9) ? a_dst : a_src;
    float s = 0.f;
    for (int c = lane; c < HD; c += 32) s += vec[c] * coef[c];
#pragma unroll
    for (int off = 16; off > 0; off >>= 1) s += __shfl_down(s, off, 32);
    if (lane == 0) sDot[dp] = s;
  }
  __syncthreads();

  if (tid == 0) {
    float e[NNODE];
    float s0 = sDot[9];
    float mx = -1e30f;
    for (int j = 0; j < NNODE; ++j) {
      bool ok = true;
      if (j > 0) {
        const unsigned char* pm = attr_mask + ((size_t)g * NATTR + (j - 1)) * TA;
        ok = (pm[0] == 0) || (pm[1] == 0) || (pm[2] == 0) || (pm[3] == 0);
      }
      float v = s0 + sDot[j];
      v = v > 0.f ? v : 0.2f * v;          // leaky_relu(0.2)
      e[j] = ok ? v : -1e9f;
      mx = fmaxf(mx, e[j]);
    }
    float sum = 0.f;
    for (int j = 0; j < NNODE; ++j) { e[j] = __expf(e[j] - mx); sum += e[j]; }
    float inv = 1.f / sum;
    for (int j = 0; j < NNODE; ++j) sAlpha[j] = e[j] * inv;
  }
  __syncthreads();

  for (int c = tid; c < HD; c += 256) {
    float acc = 0.f;
#pragma unroll
    for (int j = 0; j < NNODE; ++j) acc += sAlpha[j] * base[(size_t)j * HD + c];
    out[(size_t)g * HD + c] = acc > 0.f ? acc : (__expf(acc) - 1.f);  // elu
  }
}

// ---------------------------------------------------------------------------
extern "C" void kernel_launch(void* const* d_in, const int* in_sizes, int n_in,
                              void* d_out, int out_size, void* d_ws, size_t ws_size,
                              hipStream_t stream) {
  (void)in_sizes; (void)n_in; (void)out_size; (void)ws_size;
  const int* etok = (const int*)d_in[0];
  const int* elen = (const int*)d_in[1];
  const int* atok = (const int*)d_in[3];
  const int* alen = (const int*)d_in[4];
  const unsigned char* amask = (const unsigned char*)d_in[5];
  const float* emb   = (const float*)d_in[6];
  const float* WihE  = (const float*)d_in[7];
  const float* WhhE  = (const float*)d_in[8];
  const float* bihE  = (const float*)d_in[9];
  const float* bhhE  = (const float*)d_in[10];
  const float* WihA  = (const float*)d_in[11];
  const float* WhhA  = (const float*)d_in[12];
  const float* bihA  = (const float*)d_in[13];
  const float* bhhA  = (const float*)d_in[14];
  const float* Wfc   = (const float*)d_in[15];
  const float* bfc   = (const float*)d_in[16];
  const float* Wg    = (const float*)d_in[17];
  const float* a_src = (const float*)d_in[18];
  const float* a_dst = (const float*)d_in[19];
  float* out = (float*)d_out;

  // workspace carve-out (256B aligned slabs)
  char* p = (char*)d_ws;
  auto carve = [&](size_t bytes) {
    char* r = p; p += (bytes + 255) & ~(size_t)255; return r;
  };
  const size_t pkGateB = (size_t)NGT * KC_W * 512 * 2;
  unsigned short* pkWihE = (unsigned short*)carve(pkGateB);
  unsigned short* pkWhhE = (unsigned short*)carve(pkGateB);
  unsigned short* pkWihA = (unsigned short*)carve(pkGateB);
  unsigned short* pkWhhA = (unsigned short*)carve(pkGateB);
  unsigned short* pkWfc  = (unsigned short*)carve((size_t)HDT * KC_W * 512 * 2);
  unsigned short* pkWg   = (unsigned short*)carve((size_t)HDT * KC_H * 512 * 2);
  unsigned short* hEnt   = (unsigned short*)carve((size_t)G  * KP * 2);
  unsigned short* hAttr  = (unsigned short*)carve((size_t)GA * KP * 2);
  unsigned short* nodes  = (unsigned short*)carve((size_t)G * NNODE * HD * 2);
  float*          hgBuf  = (float*)carve((size_t)G * NNODE * HD * 4);

  // 1) pack weights to bf16 WMMA-B fragment layout
  const int gateElems = NGT * KC_W * 512;
  const int fcElems   = HDT * KC_W * 512;
  const int wgElems   = HDT * KC_H * 512;
  pack_b_kernel<<<(gateElems + 255) / 256, 256, 0, stream>>>(WihE, pkWihE, NG, WD, NGT, KC_W);
  pack_b_kernel<<<(gateElems + 255) / 256, 256, 0, stream>>>(WhhE, pkWhhE, NG, WD, NGT, KC_W);
  pack_b_kernel<<<(gateElems + 255) / 256, 256, 0, stream>>>(WihA, pkWihA, NG, WD, NGT, KC_W);
  pack_b_kernel<<<(gateElems + 255) / 256, 256, 0, stream>>>(WhhA, pkWhhA, NG, WD, NGT, KC_W);
  pack_b_kernel<<<(fcElems   + 255) / 256, 256, 0, stream>>>(Wfc,  pkWfc,  HD, WD, HDT, KC_W);
  pack_b_kernel<<<(wgElems   + 255) / 256, 256, 0, stream>>>(Wg,   pkWg,   HD, HD, HDT, KC_H);

  // 2) fused gather + GRU (entity: T=8, attribute: T=4)
  gru_wmma_kernel<<<G / 16, 128, 0, stream>>>(etok, elen, emb, pkWihE, pkWhhE,
                                              bihE, bhhE, hEnt, TE);
  gru_wmma_kernel<<<GA / 16, 128, 0, stream>>>(atok, alen, emb, pkWihA, pkWhhA,
                                               bihA, bhhA, hAttr, TA);

  // 3) FC + ReLU -> node tensor [G,9,HD] (bf16)
  fc_relu_wmma_kernel<<<G / 16, 256, 0, stream>>>(hEnt, pkWfc, bfc, nodes, 0);
  fc_relu_wmma_kernel<<<GA / 16, 256, 0, stream>>>(hAttr, pkWfc, bfc, nodes, 1);

  // 4) GAT projection hg = nodes @ Wg^T (fp32)
  gat_gemm_wmma_kernel<<<(G * NNODE) / 16, 256, 0, stream>>>(nodes, pkWg, hgBuf);

  // 5) masked attention + ELU readout for node 0 of each graph
  gat_attn_kernel<<<G, 256, 0, stream>>>(hgBuf, a_src, a_dst, amask, out);
}
